// SparseTransformerDecoder_69020124446906
// MI455X (gfx1250) — matrix operand order repaired and checked by reading, not compiled, not run on hardware
//
#include <hip/hip_runtime.h>
#include <hip/hip_bf16.h>

// ---------------------------------------------------------------------------
// Sparse transformer decoder forward, CDNA5 (gfx1250) WMMA implementation.
// B=1, S=2048, D=512, H=8, DH=64, L=6, MLP=2048, VOCAB=256, BLOCK=128, C=32
// All activations between matmuls kept in f16 (WMMA operand type); f32
// residual stream; f32 accumulation everywhere.
// ---------------------------------------------------------------------------

#define S_LEN  2048
#define D_MOD  512
#define N_HEAD 8
#define D_HEAD 64
#define N_LAYR 6
#define D_MLP  2048
#define N_VOC  256
#define QKV_W  1536           // 3 * H * DH
#define NEG_INF (-1e9f)

typedef __attribute__((ext_vector_type(16))) _Float16 v16h;
typedef __attribute__((ext_vector_type(8)))  _Float16 v8h;
typedef __attribute__((ext_vector_type(8)))  float    v8f;

union AF {
    v16h v;
    v8h  h[2];
    _Float16 e[16];
};

// A-fragment (16x32 f16); per-lane layout:
//   lane<16 : M=lane,    e[0..7]=K(k0+0..7),  e[8..15]=K(k0+16..23)
//   lane>=16: M=lane-16, e[0..7]=K(k0+8..15), e[8..15]=K(k0+24..31)
__device__ inline v16h load_a_frag_f16(const _Float16* __restrict__ rowp,
                                       int k0, int hi) {
    AF a;
    a.h[0] = *(const v8h*)(rowp + k0 + hi * 8);
    a.h[1] = *(const v8h*)(rowp + k0 + 16 + hi * 8);
    return a.v;
}

// B-fragment (32x16 f16) from a [N, Kd] row-major f16 source; lane's 16
// K-values: e[0..7]=K(k0+hi*8+0..7), e[8..15]=K(k0+16+hi*8+0..7).
__device__ inline v16h load_b_frag_f16(const _Float16* __restrict__ btrow,
                                       int k0, int hi) {
    AF b;
    b.h[0] = *(const v8h*)(btrow + k0 + hi * 8);
    b.h[1] = *(const v8h*)(btrow + k0 + 16 + hi * 8);
    return b.v;
}

__device__ inline float gelu_tanh(float x) {
    float x3 = x * x * x;
    return 0.5f * x * (1.0f + tanhf(0.7978845608028654f * (x + 0.044715f * x3)));
}

// ---------------------------------------------------------------------------
// Weight convert: src f32 [K, N] row-major -> dst f16 transposed [N, K].
// ---------------------------------------------------------------------------
__global__ __launch_bounds__(256)
void convtr_kernel(const float* __restrict__ src, _Float16* __restrict__ dst,
                   int K, int N, float scale) {
    int i = blockIdx.x * 256 + threadIdx.x;
    if (i >= K * N) return;
    int k = i / N;
    int n = i - k * N;
    dst[(size_t)n * K + k] = (_Float16)(src[i] * scale);
}

// ---------------------------------------------------------------------------
// Embedding: x = embed[shift(tokens)] + pos; key-pad mask from raw tokens.
// ---------------------------------------------------------------------------
__global__ __launch_bounds__(256)
void embed_kernel(const int* __restrict__ tokens,
                  const float* __restrict__ emb,
                  const float* __restrict__ pos,
                  float* __restrict__ x, float* __restrict__ padf) {
    int s = blockIdx.x;
    int tid = threadIdx.x;
    int tok = (s == 0) ? 0 : tokens[s - 1];
#pragma unroll
    for (int i = 0; i < 2; ++i) {
        int d = tid + i * 256;
        x[(size_t)s * D_MOD + d] =
            emb[(size_t)tok * D_MOD + d] + pos[(size_t)s * D_MOD + d];
    }
    if (tid == 0) padf[s] = (tokens[s] > 0) ? 1.0f : 0.0f;
}

// ---------------------------------------------------------------------------
// LayerNorm over D=512, one workgroup (256 threads) per row; f16 output
// (next consumer is always a WMMA GEMM A-operand).
// ---------------------------------------------------------------------------
__global__ __launch_bounds__(256)
void ln_kernel(const float* __restrict__ x, const float* __restrict__ scale,
               const float* __restrict__ bias, _Float16* __restrict__ y) {
    __shared__ float red[256];
    int row = blockIdx.x;
    int tid = threadIdx.x;
    const float* xr = x + (size_t)row * D_MOD;
    float a0 = xr[tid], a1 = xr[tid + 256];
    red[tid] = a0 + a1;
    __syncthreads();
#pragma unroll
    for (int st = 128; st > 0; st >>= 1) {
        if (tid < st) red[tid] += red[tid + st];
        __syncthreads();
    }
    float mu = red[0] * (1.0f / 512.0f);
    __syncthreads();
    float d0 = a0 - mu, d1 = a1 - mu;
    red[tid] = d0 * d0 + d1 * d1;
    __syncthreads();
#pragma unroll
    for (int st = 128; st > 0; st >>= 1) {
        if (tid < st) red[tid] += red[tid + st];
        __syncthreads();
    }
    float rstd = rsqrtf(red[0] * (1.0f / 512.0f) + 1e-6f);
    _Float16* yr = y + (size_t)row * D_MOD;
    yr[tid]       = (_Float16)(d0 * rstd * scale[tid]       + bias[tid]);
    yr[tid + 256] = (_Float16)(d1 * rstd * scale[tid + 256] + bias[tid + 256]);
}

// ---------------------------------------------------------------------------
// V transpose: qkv f16 [S, 1536] (V at col 1024+) -> vT f16 [512, S].
// Gives the attention V B-fragment contiguous key runs.
// ---------------------------------------------------------------------------
__global__ __launch_bounds__(256)
void vtr_kernel(const _Float16* __restrict__ qkv, _Float16* __restrict__ vT) {
    int i = blockIdx.x * 256 + threadIdx.x;      // over S*512
    int s = i >> 9;
    int d = i & 511;
    vT[(size_t)d * S_LEN + s] = qkv[(size_t)s * QKV_W + 1024 + d];
}

// ---------------------------------------------------------------------------
// WMMA GEMM: out[M,N] = act( A[M,Kd] (f16) x BT[N,Kd]^T (f16) + bias ) [+resid]
// 128 threads = 4 waves; each wave owns a 32(M) x 64(N) tile:
// 8 WMMAs per K-step against 4 A-fragment + 8 B-fragment b128 loads.
// Output either f32 (Cf, with optional residual) or f16 (Ch).
// ---------------------------------------------------------------------------
__global__ __launch_bounds__(128)
void gemm_kernel(const _Float16* __restrict__ A, const _Float16* __restrict__ BT,
                 const float* __restrict__ bias, const float* __restrict__ resid,
                 float* __restrict__ Cf, _Float16* __restrict__ Ch,
                 int M, int N, int Kd, int do_gelu) {
    int wave = threadIdx.x >> 5;
    int lane = threadIdx.x & 31;
    int mloc = lane & 15;
    int hi   = lane >> 4;
    int m0 = blockIdx.x * 32;
    int n0 = (blockIdx.y * 4 + wave) * 64;
    if (m0 >= M || n0 >= N) return;

    const _Float16* arow0 = A + (size_t)(m0 + mloc) * Kd;
    const _Float16* arow1 = arow0 + (size_t)16 * Kd;
    v8f zer = {0.f,0.f,0.f,0.f,0.f,0.f,0.f,0.f};
    v8f acc[2][4] = {{zer, zer, zer, zer}, {zer, zer, zer, zer}};

    for (int k0 = 0; k0 < Kd; k0 += 32) {
        v16h af0 = load_a_frag_f16(arow0, k0, hi);
        v16h af1 = load_a_frag_f16(arow1, k0, hi);
#pragma unroll
        for (int nt = 0; nt < 4; ++nt) {
            const _Float16* btrow = BT + (size_t)(n0 + nt * 16 + mloc) * Kd;
            __builtin_prefetch(btrow + k0 + 64, 0, 1);   // global_prefetch_b8
            v16h bf = load_b_frag_f16(btrow, k0, hi);
            acc[0][nt] = __builtin_amdgcn_wmma_f32_16x16x32_f16(
                false, af0, false, bf, (short)0, acc[0][nt], false, false);
            acc[1][nt] = __builtin_amdgcn_wmma_f32_16x16x32_f16(
                false, af1, false, bf, (short)0, acc[1][nt], false, false);
        }
    }

#pragma unroll
    for (int mi = 0; mi < 2; ++mi) {
#pragma unroll
        for (int nt = 0; nt < 4; ++nt) {
#pragma unroll
            for (int r = 0; r < 8; ++r) {
                int row = m0 + mi * 16 + r + hi * 8;
                int col = n0 + nt * 16 + mloc;
                float v = acc[mi][nt][r];
                if (bias)    v += bias[col];
                if (do_gelu) v = gelu_tanh(v);
                if (Ch) {
                    Ch[(size_t)row * N + col] = (_Float16)v;
                } else {
                    if (resid) v += resid[(size_t)row * N + col];
                    Cf[(size_t)row * N + col] = v;
                }
            }
        }
    }
}

// ---------------------------------------------------------------------------
// Block-sparse flash attention. One wave per (16-query tile, head).
// Sparse pattern: causal & (same 128-block | last-32 columns of each block).
// Key tiles of 32: one stripe tile (cols kb*128+96..127) per earlier block,
// plus the diagonal-block tiles up to the causal frontier.
// All operands f16 (Q/K from qkv, V from vT), f32 accumulate, f16 output.
// ---------------------------------------------------------------------------
__global__ __launch_bounds__(32)
void attn_kernel(const _Float16* __restrict__ qkv,
                 const _Float16* __restrict__ vT,
                 const float* __restrict__ padf,
                 _Float16* __restrict__ o) {
    __shared__ _Float16 plds[16 * 32];
    int lane = threadIdx.x;
    int mloc = lane & 15;
    int hi   = lane >> 4;
    int q0 = blockIdx.x * 16;
    int h  = blockIdx.y;
    int qb = q0 >> 7;                      // 128-block index of this query tile

    const _Float16* qrow = qkv + (size_t)(q0 + mloc) * QKV_W + h * D_HEAD;
    v16h qf0 = load_a_frag_f16(qrow, 0, hi);
    v16h qf1 = load_a_frag_f16(qrow, 32, hi);

    float mrun[8], lrun[8];
    v8f zer = {0.f,0.f,0.f,0.f,0.f,0.f,0.f,0.f};
    v8f oacc[4] = {zer, zer, zer, zer};
#pragma unroll
    for (int r = 0; r < 8; ++r) { mrun[r] = -3.0e38f; lrun[r] = 0.0f; }

    int ndiag  = ((q0 & 127) >> 5) + 1;    // diagonal-block 32-key tiles
    int ntiles = qb + ndiag;               // + one stripe tile per earlier block

    for (int t = 0; t < ntiles; ++t) {
        int c0 = (t < qb) ? (t * 128 + 96) : (qb * 128 + (t - qb) * 32);

        // --- scores: Q(16x64) x K^T -> 16x32 (two 16x16 WMMA columns) -----
        v8f sc[2] = {zer, zer};
#pragma unroll
        for (int j = 0; j < 2; ++j) {
            const _Float16* krow =
                qkv + (size_t)(c0 + j * 16 + mloc) * QKV_W + 512 + h * D_HEAD;
            v16h b0 = load_b_frag_f16(krow, 0, hi);    // K dims 0..31
            v16h b1 = load_b_frag_f16(krow, 32, hi);   // K dims 32..63
            sc[j] = __builtin_amdgcn_wmma_f32_16x16x32_f16(
                false, qf0, false, b0, (short)0, sc[j], false, false);
            sc[j] = __builtin_amdgcn_wmma_f32_16x16x32_f16(
                false, qf1, false, b1, (short)0, sc[j], false, false);
        }

        float pad0 = padf[c0 + mloc];
        float pad1 = padf[c0 + 16 + mloc];
        int col0 = c0 + mloc;
        int col1 = c0 + 16 + mloc;

        // --- online softmax update (16-lane butterflies, wave32) ----------
#pragma unroll
        for (int r = 0; r < 8; ++r) {
            int rowg = q0 + r + hi * 8;
            float s0 = sc[0][r], s1 = sc[1][r];
            s0 = ((pad0 > 0.5f) && (col0 <= rowg)) ? s0 : NEG_INF;
            s1 = ((pad1 > 0.5f) && (col1 <= rowg)) ? s1 : NEG_INF;
            float mx = fmaxf(s0, s1);
#pragma unroll
            for (int d = 8; d >= 1; d >>= 1)
                mx = fmaxf(mx, __shfl_xor(mx, d, 32));
            float mnew = fmaxf(mrun[r], mx);
            float p0 = __expf(s0 - mnew);
            float p1 = __expf(s1 - mnew);
            float ps = p0 + p1;
#pragma unroll
            for (int d = 8; d >= 1; d >>= 1)
                ps += __shfl_xor(ps, d, 32);
            float alpha = __expf(mrun[r] - mnew);
            lrun[r] = lrun[r] * alpha + ps;
            mrun[r] = mnew;
#pragma unroll
            for (int nt = 0; nt < 4; ++nt) oacc[nt][r] *= alpha;
            plds[(r + hi * 8) * 32 + mloc]      = (_Float16)p0;
            plds[(r + hi * 8) * 32 + 16 + mloc] = (_Float16)p1;
        }
        __syncthreads();

        // --- P fragment (A operand 16x32) from LDS ------------------------
        AF pa;
        const _Float16* prow = plds + mloc * 32;
        pa.h[0] = *(const v8h*)(prow + hi * 8);         // K = hi*8 + 0..7
        pa.h[1] = *(const v8h*)(prow + 16 + hi * 8);    // K = 16 + hi*8 + 0..7

        // --- O += P(16x32) x V(32x64): V B-frag contiguous from vT --------
#pragma unroll
        for (int nt = 0; nt < 4; ++nt) {
            int dim = nt * 16 + mloc;
            const _Float16* vrow = vT + (size_t)(h * D_HEAD + dim) * S_LEN;
            v16h vb = load_b_frag_f16(vrow, c0, hi);    // keys c0+hi*8+0..7, c0+16+hi*8+0..7
            oacc[nt] = __builtin_amdgcn_wmma_f32_16x16x32_f16(
                false, pa.v, false, vb, (short)0, oacc[nt], false, false);
        }
        __syncthreads();
    }

    // --- normalize and store o[s, h*64 + dim] (f16) ------------------------
#pragma unroll
    for (int r = 0; r < 8; ++r) {
        float invl = 1.0f / lrun[r];
        int row = q0 + r + hi * 8;
#pragma unroll
        for (int nt = 0; nt < 4; ++nt) {
            o[(size_t)row * D_MOD + h * D_HEAD + nt * 16 + mloc] =
                (_Float16)(oacc[nt][r] * invl);
        }
    }
}

// ---------------------------------------------------------------------------
// Host orchestration.
// ---------------------------------------------------------------------------
extern "C" void kernel_launch(void* const* d_in, const int* in_sizes, int n_in,
                              void* d_out, int out_size, void* d_ws, size_t ws_size,
                              hipStream_t stream) {
    (void)in_sizes; (void)n_in; (void)out_size; (void)ws_size;

    const int*   tokens    = (const int*)d_in[0];
    const float* emb       = (const float*)d_in[1];
    const float* pos       = (const float*)d_in[2];
    const float* ln1_scale = (const float*)d_in[3];
    const float* ln1_bias  = (const float*)d_in[4];
    const float* wq        = (const float*)d_in[5];
    const float* wk        = (const float*)d_in[6];
    const float* wv        = (const float*)d_in[7];
    const float* wo        = (const float*)d_in[8];
    const float* ln2_scale = (const float*)d_in[9];
    const float* ln2_bias  = (const float*)d_in[10];
    const float* w1        = (const float*)d_in[11];
    const float* b1        = (const float*)d_in[12];
    const float* w2        = (const float*)d_in[13];
    const float* b2        = (const float*)d_in[14];
    const float* lnf_scale = (const float*)d_in[15];
    const float* lnf_bias  = (const float*)d_in[16];
    const float* wout      = (const float*)d_in[17];
    const float* bout      = (const float*)d_in[18];
    float* out = (float*)d_out;

    // --- workspace bump allocator (256B aligned) ---------------------------
    char* ws = (char*)d_ws;
    size_t off = 0;
    auto alloc = [&](size_t bytes) -> void* {
        void* p = ws + off;
        off = (off + bytes + 255) & ~(size_t)255;
        return p;
    };
    float*    xf   = (float*)alloc((size_t)S_LEN * D_MOD * 4);      // residual
    _Float16* hbh  = (_Float16*)alloc((size_t)S_LEN * D_MOD * 2);   // LN out
    _Float16* qkvh = (_Float16*)alloc((size_t)S_LEN * QKV_W * 2);
    _Float16* vTh  = (_Float16*)alloc((size_t)D_MOD * S_LEN * 2);
    _Float16* obh  = (_Float16*)alloc((size_t)S_LEN * D_MOD * 2);
    _Float16* mbh  = (_Float16*)alloc((size_t)S_LEN * D_MLP * 2);
    float*    padf = (float*)alloc((size_t)S_LEN * 4);
    _Float16* wqkvT = (_Float16*)alloc((size_t)N_LAYR * QKV_W * D_MOD * 2);
    _Float16* woT   = (_Float16*)alloc((size_t)N_LAYR * D_MOD * D_MOD * 2);
    _Float16* w1T   = (_Float16*)alloc((size_t)N_LAYR * D_MLP * D_MOD * 2);
    _Float16* w2T   = (_Float16*)alloc((size_t)N_LAYR * D_MOD * D_MLP * 2);
    _Float16* woutT = (_Float16*)alloc((size_t)N_VOC * D_MOD * 2);

    // --- weight conversion (f32 -> transposed f16; 1/sqrt(DH) folded into Q)
    const int blk512x512  = (D_MOD * D_MOD + 255) / 256;
    const int blk512x2048 = (D_MOD * D_MLP + 255) / 256;
    for (int l = 0; l < N_LAYR; ++l) {
        const size_t wsz = (size_t)D_MOD * D_MOD;
        _Float16* lq = wqkvT + (size_t)l * QKV_W * D_MOD;
        convtr_kernel<<<blk512x512, 256, 0, stream>>>(
            wq + (size_t)l * wsz, lq, D_MOD, D_MOD, 0.125f);
        convtr_kernel<<<blk512x512, 256, 0, stream>>>(
            wk + (size_t)l * wsz, lq + (size_t)512 * D_MOD, D_MOD, D_MOD, 1.0f);
        convtr_kernel<<<blk512x512, 256, 0, stream>>>(
            wv + (size_t)l * wsz, lq + (size_t)1024 * D_MOD, D_MOD, D_MOD, 1.0f);
        convtr_kernel<<<blk512x512, 256, 0, stream>>>(
            wo + (size_t)l * wsz, woT + (size_t)l * wsz, D_MOD, D_MOD, 1.0f);
        convtr_kernel<<<blk512x2048, 256, 0, stream>>>(
            w1 + (size_t)l * D_MOD * D_MLP, w1T + (size_t)l * D_MLP * D_MOD,
            D_MOD, D_MLP, 1.0f);
        convtr_kernel<<<blk512x2048, 256, 0, stream>>>(
            w2 + (size_t)l * D_MLP * D_MOD, w2T + (size_t)l * D_MOD * D_MLP,
            D_MLP, D_MOD, 1.0f);
    }
    convtr_kernel<<<(D_MOD * N_VOC + 255) / 256, 256, 0, stream>>>(
        wout, woutT, D_MOD, N_VOC, 1.0f);

    // --- embedding + pad mask ---------------------------------------------
    embed_kernel<<<S_LEN, 256, 0, stream>>>(tokens, emb, pos, xf, padf);

    for (int l = 0; l < N_LAYR; ++l) {
        // LN1 -> f16
        ln_kernel<<<S_LEN, 256, 0, stream>>>(
            xf, ln1_scale + l * D_MOD, ln1_bias + l * D_MOD, hbh);
        // fused QKV projection [2048,512] x [512,1536] -> f16
        gemm_kernel<<<dim3(S_LEN / 32, QKV_W / 256), 128, 0, stream>>>(
            hbh, wqkvT + (size_t)l * QKV_W * D_MOD, nullptr, nullptr,
            nullptr, qkvh, S_LEN, QKV_W, D_MOD, 0);
        // V transpose for contiguous B-fragments
        vtr_kernel<<<(S_LEN * D_MOD) / 256, 256, 0, stream>>>(qkvh, vTh);
        // block-sparse flash attention -> f16
        attn_kernel<<<dim3(S_LEN / 16, N_HEAD), 32, 0, stream>>>(
            qkvh, vTh, padf, obh);
        // output projection + residual -> f32 residual stream
        gemm_kernel<<<dim3(S_LEN / 32, D_MOD / 256), 128, 0, stream>>>(
            obh, woT + (size_t)l * D_MOD * D_MOD, nullptr, xf,
            xf, nullptr, S_LEN, D_MOD, D_MOD, 0);
        // LN2 -> f16
        ln_kernel<<<S_LEN, 256, 0, stream>>>(
            xf, ln2_scale + l * D_MOD, ln2_bias + l * D_MOD, hbh);
        // MLP up + GELU -> f16
        gemm_kernel<<<dim3(S_LEN / 32, D_MLP / 256), 128, 0, stream>>>(
            hbh, w1T + (size_t)l * D_MLP * D_MOD, b1 + l * D_MLP, nullptr,
            nullptr, mbh, S_LEN, D_MLP, D_MOD, 1);
        // MLP down + residual -> f32 residual stream
        gemm_kernel<<<dim3(S_LEN / 32, D_MOD / 256), 128, 0, stream>>>(
            mbh, w2T + (size_t)l * D_MOD * D_MLP, b2 + l * D_MOD, xf,
            xf, nullptr, S_LEN, D_MOD, D_MLP, 0);
    }

    // final LN + logits
    ln_kernel<<<S_LEN, 256, 0, stream>>>(xf, lnf_scale, lnf_bias, hbh);
    gemm_kernel<<<dim3(S_LEN / 32, N_VOC / 256), 128, 0, stream>>>(
        hbh, woutT, bout, nullptr, out, nullptr, S_LEN, N_VOC, D_MOD, 0);
}
